// EdgePred_GPrompt_326417514918
// MI455X (gfx1250) — matrix operand order repaired and checked by reading, not compiled, not run on hardware
//
#include <hip/hip_runtime.h>
#include <hip/hip_bf16.h>

typedef __attribute__((ext_vector_type(16))) _Float16 v16h;
typedef __attribute__((ext_vector_type(8)))  _Float16 v8h;
typedef __attribute__((ext_vector_type(4)))  _Float16 v4h;
typedef __attribute__((ext_vector_type(8)))  float    v8f;
typedef int v4i_vec __attribute__((vector_size(16)));   // matches builtin's pointee type

// gfx1250 async global->LDS copy (ASYNCcnt-tracked), with safe fallback.
#if defined(__gfx1250__) && __has_builtin(__builtin_amdgcn_global_load_async_to_lds_b128)
#define USE_ASYNC_LDS 1
#else
#define USE_ASYNC_LDS 0
#endif

#if USE_ASYNC_LDS
static __device__ inline __attribute__((address_space(3))) v4i_vec* to_lds(void* p) {
    return (__attribute__((address_space(3))) v4i_vec*)p;        // addrspacecast generic->LDS
}
static __device__ inline __attribute__((address_space(1))) v4i_vec* to_global(const void* p) {
    return (__attribute__((address_space(1))) v4i_vec*)(unsigned long long)p;  // value cast (AS0/AS1 same repr)
}
static __device__ inline void wait_async_zero() {
#if __has_builtin(__builtin_amdgcn_s_wait_asynccnt)
    __builtin_amdgcn_s_wait_asynccnt(0);
#else
    asm volatile("s_wait_asynccnt 0" ::: "memory");
#endif
}
#endif

// ---------------------------------------------------------------- utilities

__global__ void fill_f32_kernel(float* p, float v, long n) {
    long i = (long)blockIdx.x * blockDim.x + threadIdx.x;
    long stride = (long)gridDim.x * blockDim.x;
    for (; i < n; i += stride) p[i] = v;
}

// GCN degree: deg[v] += 1 per edge (deg pre-initialized to 1.0 for self loop)
__global__ void deg_gcn_kernel(const int* __restrict__ v, float* deg, int E) {
    int e = blockIdx.x * blockDim.x + threadIdx.x;
    if (e < E) atomicAdd(&deg[v[e]], 1.0f);
}

// Message-passing degree: bidirectional, skip self edges
__global__ void deg_mp_kernel(const int* __restrict__ u, const int* __restrict__ v,
                              float* deg, int E) {
    int e = blockIdx.x * blockDim.x + threadIdx.x;
    if (e < E) {
        int a = u[e], b = v[e];
        if (a != b) { atomicAdd(&deg[b], 1.0f); atomicAdd(&deg[a], 1.0f); }
    }
}

// dis = deg > 0 ? rsqrt(max(deg,1)) : 0   (in place)
__global__ void finalize_dis_kernel(float* p, int n) {
    int i = blockIdx.x * blockDim.x + threadIdx.x;
    if (i < n) {
        float d = p[i];
        p[i] = (d > 0.0f) ? rsqrtf(fmaxf(d, 1.0f)) : 0.0f;
    }
}

// ---------------------------------------------------------------- WMMA GEMM
// C[M x Nout] = A[M x K] * W[K x Nout] (+ bias), RELU optionally applied to A.
// Block: 256 threads = 8 waves. Workgroup tile: 128 x TILE_N.
// Waves arranged 4 (row groups of 32) x 2 (col groups of TILE_N/2).
// Each wave: 2 x (TILE_N/32) accumulators of v_wmma_f32_16x16x32_f16.
// A tile: DMA'd f32 into LDS via GLOBAL_LOAD_ASYNC_TO_LDS_B128 (ASYNCcnt),
//         converted to f16 (+ReLU) during per-wave fragment assembly.
// Row guard via index clamp (padded A rows only feed padded C rows, never stored).
template<int TILE_N, int RELU>
__launch_bounds__(256)
__global__ void gemm_wmma_f16_kernel(const float* __restrict__ A,
                                     const float* __restrict__ W,
                                     const float* __restrict__ bias,
                                     float* __restrict__ C,
                                     int M, int K, int Nout) {
    constexpr int NJ = TILE_N / 32;          // col subtiles per wave (2 or 4)
#if USE_ASYNC_LDS
    __shared__ __align__(16) float    ldsAf[128 * 32];      // [row][k]  f32, async-filled
#else
    __shared__ __align__(16) _Float16 ldsA [128 * 32];      // [row][k]  f16
#endif
    __shared__ __align__(16) _Float16 ldsB[TILE_N * 32];    // [n][k] (transposed) f16

    const int t    = threadIdx.x;
    const int lane = t & 31;
    const int wave = t >> 5;
    const int wm   = wave & 3;       // row group: 32 rows each
    const int wn   = wave >> 2;      // col group: TILE_N/2 cols each
    const int half = lane >> 4;      // 0 | 1
    const int l16  = lane & 15;

    const int m0 = blockIdx.x * 128;
    const int n0 = blockIdx.y * TILE_N;
    const int Mc = M - 1;

    v8f acc[2][NJ] = {};

    for (int k0 = 0; k0 < K; k0 += 32) {
        // ---- Stage A tile (128 rows x 32 k) -------------------------------
        #pragma unroll
        for (int i = 0; i < 4; ++i) {
            int idx = t + i * 256;               // 0..1023 16B chunks
            int r = idx >> 3, c4 = (idx & 7) * 4;
            int gr = m0 + r; if (gr > Mc) gr = Mc;
#if USE_ASYNC_LDS
            __builtin_amdgcn_global_load_async_to_lds_b128(
                to_global(A + (long)gr * K + (k0 + c4)),
                to_lds(&ldsAf[r * 32 + c4]), 0, 0);
#else
            float4 val = *(const float4*)(A + (long)gr * K + (k0 + c4));
            if (RELU) {
                val.x = val.x > 0.0f ? val.x : 0.0f;
                val.y = val.y > 0.0f ? val.y : 0.0f;
                val.z = val.z > 0.0f ? val.z : 0.0f;
                val.w = val.w > 0.0f ? val.w : 0.0f;
            }
            v4h hv;
            hv[0] = (_Float16)val.x; hv[1] = (_Float16)val.y;
            hv[2] = (_Float16)val.z; hv[3] = (_Float16)val.w;
            *(v4h*)&ldsA[r * 32 + c4] = hv;
#endif
        }
        // ---- Stage B tile transposed: coalesced float4 load, b16 scatter --
        #pragma unroll
        for (int i = 0; i < TILE_N / 32; ++i) {
            int idx = t + i * 256;               // 0..TILE_N*8-1 float4 chunks
            int k  = idx / (TILE_N / 4);
            int n4 = (idx % (TILE_N / 4)) * 4;
            float4 val = *(const float4*)(W + (long)(k0 + k) * Nout + (n0 + n4));
            ldsB[(n4 + 0) * 32 + k] = (_Float16)val.x;
            ldsB[(n4 + 1) * 32 + k] = (_Float16)val.y;
            ldsB[(n4 + 2) * 32 + k] = (_Float16)val.z;
            ldsB[(n4 + 3) * 32 + k] = (_Float16)val.w;
        }
#if USE_ASYNC_LDS
        wait_async_zero();
#endif
        __syncthreads();

        // ---- A fragments (ISA 16-bit A 16x32 layout) ----------------------
        // lanes 0-15: row M=l16, K 0-7 & 16-23 ; lanes 16-31: K 8-15 & 24-31
        v16h aF[2], bF[NJ];
        #pragma unroll
        for (int i = 0; i < 2; ++i) {
#if USE_ASYNC_LDS
            const float* ap = &ldsAf[(wm * 32 + i * 16 + l16) * 32 + half * 8];
            float4 q0 = *(const float4*)(ap);
            float4 q1 = *(const float4*)(ap + 4);
            float4 q2 = *(const float4*)(ap + 16);
            float4 q3 = *(const float4*)(ap + 20);
            float qa[16] = { q0.x, q0.y, q0.z, q0.w, q1.x, q1.y, q1.z, q1.w,
                             q2.x, q2.y, q2.z, q2.w, q3.x, q3.y, q3.z, q3.w };
            #pragma unroll
            for (int e = 0; e < 16; ++e) {
                float f = qa[e];
                if (RELU) f = f > 0.0f ? f : 0.0f;
                aF[i][e] = (_Float16)f;
            }
#else
            const _Float16* ap = &ldsA[(wm * 32 + i * 16 + l16) * 32 + half * 8];
            v8h lo = *(const v8h*)ap;
            v8h hi = *(const v8h*)(ap + 16);
            #pragma unroll
            for (int e = 0; e < 8; ++e) { aF[i][e] = lo[e]; aF[i][8 + e] = hi[e]; }
#endif
        }
        // ---- B fragments (16-bit B 32x16): lane col N=l16; lanes 0-15 K 0-7 & 16-23,
        // lanes 16-31 K 8-15 & 24-31 (matches A)
        #pragma unroll
        for (int j = 0; j < NJ; ++j) {
            const _Float16* bp = &ldsB[(wn * (TILE_N / 2) + j * 16 + l16) * 32 + half * 8];
            v8h lo = *(const v8h*)bp;
            v8h hi = *(const v8h*)(bp + 16);
            #pragma unroll
            for (int e = 0; e < 8; ++e) { bF[j][e] = lo[e]; bF[j][8 + e] = hi[e]; }
        }

        #pragma unroll
        for (int i = 0; i < 2; ++i)
            #pragma unroll
            for (int j = 0; j < NJ; ++j)
                acc[i][j] = __builtin_amdgcn_wmma_f32_16x16x32_f16(
                    false, aF[i], false, bF[j], (short)0, acc[i][j], false, false);
        __syncthreads();
    }

    // Epilogue: C layout — VGPR e -> row = e + 8*half, col = l16
    #pragma unroll
    for (int i = 0; i < 2; ++i) {
        int baseM = m0 + wm * 32 + i * 16 + half * 8;
        #pragma unroll
        for (int j = 0; j < NJ; ++j) {
            int gc = n0 + wn * (TILE_N / 2) + j * 16 + l16;
            float bv = bias ? bias[gc] : 0.0f;
            #pragma unroll
            for (int e = 0; e < 8; ++e) {
                int gr = baseM + e;
                if (gr < M) C[(long)gr * Nout + gc] = acc[i][j][e] + bv;
            }
        }
    }
}

// ------------------------------------------------------- GCN edge aggregation

// out[i][c] = dis[i]^2 * h[i][c] + bias[c]   (self-loop term + bias), Cdim=256
__global__ void agg_init_kernel(const float* __restrict__ h,
                                const float* __restrict__ dis,
                                const float* __restrict__ bias,
                                float* __restrict__ out, long total) {
    long i = (long)blockIdx.x * blockDim.x + threadIdx.x;
    if (i < total) {
        int node = (int)(i >> 8);     // Cdim = 256
        int c    = (int)(i & 255);
        float d = dis[node];
        out[i] = d * d * h[i] + bias[c];
    }
}

// out[v] += dis[u]*dis[v]*h[u], 256 channels, 32 threads/edge (8 ch each)
__global__ void agg_edges_kernel(const int* __restrict__ u, const int* __restrict__ v,
                                 const float* __restrict__ dis,
                                 const float* __restrict__ h,
                                 float* __restrict__ out, int E) {
    long g = (long)blockIdx.x * blockDim.x + threadIdx.x;
    int e  = (int)(g >> 5);
    int c8 = (int)(g & 31) * 8;
    if (e < E) {
        int a = u[e], b = v[e];
        float nrm = dis[a] * dis[b];
        const float* hp = h + (long)a * 256 + c8;
        float* op = out + (long)b * 256 + c8;
        float4 x0 = *(const float4*)(hp);
        float4 x1 = *(const float4*)(hp + 4);
        atomicAdd(op + 0, nrm * x0.x); atomicAdd(op + 1, nrm * x0.y);
        atomicAdd(op + 2, nrm * x0.z); atomicAdd(op + 3, nrm * x0.w);
        atomicAdd(op + 4, nrm * x1.x); atomicAdd(op + 5, nrm * x1.y);
        atomicAdd(op + 6, nrm * x1.z); atomicAdd(op + 7, nrm * x1.w);
    }
}

// One message-passing hop: y[v]+=n*x[u], y[u]+=n*x[v]; 64 ch, 8 threads/edge
__global__ void mp_edges_kernel(const int* __restrict__ u, const int* __restrict__ v,
                                const float* __restrict__ dis,
                                const float* __restrict__ x,
                                float* __restrict__ y, int E) {
    long g = (long)blockIdx.x * blockDim.x + threadIdx.x;
    int e  = (int)(g >> 3);
    int c8 = (int)(g & 7) * 8;
    if (e < E) {
        int a = u[e], b = v[e];
        if (a != b) {
            float nrm = dis[a] * dis[b];
            const float* xa = x + (long)a * 64 + c8;
            const float* xb = x + (long)b * 64 + c8;
            float4 a0 = *(const float4*)(xa), a1 = *(const float4*)(xa + 4);
            float4 b0 = *(const float4*)(xb), b1 = *(const float4*)(xb + 4);
            float* yb = y + (long)b * 64 + c8;
            float* ya = y + (long)a * 64 + c8;
            atomicAdd(yb + 0, nrm * a0.x); atomicAdd(yb + 1, nrm * a0.y);
            atomicAdd(yb + 2, nrm * a0.z); atomicAdd(yb + 3, nrm * a0.w);
            atomicAdd(yb + 4, nrm * a1.x); atomicAdd(yb + 5, nrm * a1.y);
            atomicAdd(yb + 6, nrm * a1.z); atomicAdd(yb + 7, nrm * a1.w);
            atomicAdd(ya + 0, nrm * b0.x); atomicAdd(ya + 1, nrm * b0.y);
            atomicAdd(ya + 2, nrm * b0.z); atomicAdd(ya + 3, nrm * b0.w);
            atomicAdd(ya + 4, nrm * b1.x); atomicAdd(ya + 5, nrm * b1.y);
            atomicAdd(ya + 6, nrm * b1.z); atomicAdd(ya + 7, nrm * b1.w);
        }
    }
}

// ---------------------------------------------------------------- loss

__launch_bounds__(256)
__global__ void loss_kernel(const float* __restrict__ emb,
                            const int* __restrict__ batch,
                            int B, float* __restrict__ out) {
    __shared__ float red[256];
    int i = blockIdx.x * 256 + threadIdx.x;
    float l = 0.0f;
    if (i < B) {
        int a = batch[3 * i], p = batch[3 * i + 1], n = batch[3 * i + 2];
        const float4* pa = (const float4*)(emb + (long)a * 64);
        const float4* pp = (const float4*)(emb + (long)p * 64);
        const float4* pn = (const float4*)(emb + (long)n * 64);
        float dap = 0.f, dan = 0.f, naa = 0.f, npp = 0.f, nnn = 0.f;
        #pragma unroll
        for (int k = 0; k < 16; ++k) {
            float4 va = pa[k], vp = pp[k], vn = pn[k];
            dap += va.x * vp.x + va.y * vp.y + va.z * vp.z + va.w * vp.w;
            dan += va.x * vn.x + va.y * vn.y + va.z * vn.z + va.w * vn.w;
            naa += va.x * va.x + va.y * va.y + va.z * va.z + va.w * va.w;
            npp += vp.x * vp.x + vp.y * vp.y + vp.z * vp.z + vp.w * vp.w;
            nnn += vn.x * vn.x + vn.y * vn.y + vn.z * vn.z + vn.w * vn.w;
        }
        float sa = fmaxf(sqrtf(naa), 1e-8f);
        float sp = fmaxf(sqrtf(npp), 1e-8f);
        float sn = fmaxf(sqrtf(nnn), 1e-8f);
        float cap = dap / (sa * sp);
        float can = dan / (sa * sn);
        float xs = expf(cap * 5.0f);   // 1/TEMP = 5
        float ys = expf(can * 5.0f);
        l = -logf(xs / (xs + ys));
    }
    red[threadIdx.x] = l;
    __syncthreads();
    for (int s = 128; s > 0; s >>= 1) {
        if (threadIdx.x < s) red[threadIdx.x] += red[threadIdx.x + s];
        __syncthreads();
    }
    if (threadIdx.x == 0) atomicAdd(out, red[0] / (float)B);
}

// ---------------------------------------------------------------- launch

extern "C" void kernel_launch(void* const* d_in, const int* in_sizes, int n_in,
                              void* d_out, int out_size, void* d_ws, size_t ws_size,
                              hipStream_t stream) {
    const float* x     = (const float*)d_in[0];
    const int*   ei    = (const int*)d_in[1];
    const int*   batch = (const int*)d_in[2];
    const float* W1    = (const float*)d_in[3];
    const float* b1    = (const float*)d_in[4];
    const float* W2    = (const float*)d_in[5];
    const float* b2    = (const float*)d_in[6];
    const float* Wp    = (const float*)d_in[7];
    const float* bp    = (const float*)d_in[8];

    const int H    = in_sizes[4];            // 256
    const int Dout = in_sizes[8];            // 64
    const int Din  = in_sizes[3] / H;        // 256
    const int N    = in_sizes[0] / Din;      // 100000
    const int E    = in_sizes[1] / 2;        // 800000
    const int B    = in_sizes[2] / 3;        // 100000
    const int* u = ei;
    const int* v = ei + E;

    // Workspace layout (floats)
    float* ws   = (float*)d_ws;
    float* dis1 = ws;                         // N   (GCN symmetric norm)
    float* dis2 = dis1 + N;                   // N   (message-passing norm)
    float* bufA = dis2 + N;                   // N*256
    float* bufB = bufA + (long)N * H;         // N*256
    float* embA = bufB + (long)N * H;         // N*64
    float* embB = embA + (long)N * Dout;      // N*64

    const int TB = 256;
    dim3 blk(TB);
    int edge_blocks = (E + TB - 1) / TB;

    // 1) degrees -> dis
    fill_f32_kernel<<<1024, blk, 0, stream>>>(dis1, 1.0f, (long)N);   // +1 self loop
    fill_f32_kernel<<<1024, blk, 0, stream>>>(dis2, 0.0f, (long)N);
    deg_gcn_kernel<<<edge_blocks, blk, 0, stream>>>(v, dis1, E);
    deg_mp_kernel<<<edge_blocks, blk, 0, stream>>>(u, v, dis2, E);
    finalize_dis_kernel<<<(N + TB - 1) / TB, blk, 0, stream>>>(dis1, N);
    finalize_dis_kernel<<<(N + TB - 1) / TB, blk, 0, stream>>>(dis2, N);

    dim3 gemm_grid((N + 127) / 128, H / 128);       // TILE_N = 128
    dim3 gemm_grid_p((N + 127) / 128, Dout / 64);   // TILE_N = 64
    long totalNH = (long)N * H;
    int agg_init_blocks = (int)((totalNH + TB - 1) / TB);
    int agg_edge_blocks = (int)(((long)E * 32 + TB - 1) / TB);
    int mp_edge_blocks  = (int)(((long)E * 8 + TB - 1) / TB);

    // 2) conv1: h = x@W1 -> aggregate + b1   (ReLU deferred into next GEMM load)
    gemm_wmma_f16_kernel<128, 0><<<gemm_grid, blk, 0, stream>>>(x, W1, nullptr, bufA, N, Din, H);
    agg_init_kernel<<<agg_init_blocks, blk, 0, stream>>>(bufA, dis1, b1, bufB, totalNH);
    agg_edges_kernel<<<agg_edge_blocks, blk, 0, stream>>>(u, v, dis1, bufA, bufB, E);

    // 3) conv2: h = relu(bufB)@W2 -> aggregate + b2
    gemm_wmma_f16_kernel<128, 1><<<gemm_grid, blk, 0, stream>>>(bufB, W2, nullptr, bufA, N, H, H);
    agg_init_kernel<<<agg_init_blocks, blk, 0, stream>>>(bufA, dis1, b2, bufB, totalNH);
    agg_edges_kernel<<<agg_edge_blocks, blk, 0, stream>>>(u, v, dis1, bufA, bufB, E);

    // 4) projection: emb = bufB@Wp + bp
    gemm_wmma_f16_kernel<64, 0><<<gemm_grid_p, blk, 0, stream>>>(bufB, Wp, bp, embA, N, H, Dout);

    // 5) two message-passing hops
    fill_f32_kernel<<<1024, blk, 0, stream>>>(embB, 0.0f, (long)N * Dout);
    mp_edges_kernel<<<mp_edge_blocks, blk, 0, stream>>>(u, v, dis2, embA, embB, E);
    fill_f32_kernel<<<1024, blk, 0, stream>>>(embA, 0.0f, (long)N * Dout);
    mp_edges_kernel<<<mp_edge_blocks, blk, 0, stream>>>(u, v, dis2, embB, embA, E);

    // 6) contrastive loss -> scalar mean
    fill_f32_kernel<<<1, 1, 0, stream>>>((float*)d_out, 0.0f, 1);
    loss_kernel<<<(B + TB - 1) / TB, blk, 0, stream>>>(embA, batch, B, (float*)d_out);
}
